// EncoderBlock_68650757259376
// MI455X (gfx1250) — matrix operand order repaired and checked by reading, not compiled, not run on hardware
//
#include <hip/hip_runtime.h>

#define DIMS   4096
#define SEQ    2048
#define BATCH  4
#define NHEADS 32
#define NKV    8
#define HD     128
#define FFN    11008
#define TOKENS (BATCH * SEQ)  // 8192
#define TWS    40             // LDS weight-tile row stride in ushorts (80 B, bank-friendly)

typedef unsigned short u16;
typedef unsigned int   u32;
typedef __attribute__((ext_vector_type(16))) __bf16 bf16x16;
typedef __attribute__((ext_vector_type(8)))  float  f32x8;

union FragU { u32 u[8]; bf16x16 v; };

__device__ __forceinline__ u16 f2bf(float f) {
  union { float f; u32 u; } x; x.f = f;
  u32 r = x.u + 0x7fffu + ((x.u >> 16) & 1u);   // round-to-nearest-even
  return (u16)(r >> 16);
}

__device__ __forceinline__ unsigned lds_off_of(const void* p) {
  // LDS aperture: low 32 bits of the flat address are the LDS byte offset.
  return (unsigned)(size_t)p;
}

// GLOBAL_LOAD_ASYNC_TO_LDS_B128 (GVS mode: sgpr base + 32-bit vgpr offset), ASYNCcnt-tracked.
__device__ __forceinline__ void async_ld_b128(unsigned lds_off, unsigned long long gbase,
                                              unsigned goff) {
  asm volatile("global_load_async_to_lds_b128 %0, %1, %2"
               :: "v"(lds_off), "v"(goff), "s"(gbase)
               : "memory");
}

// ---------------------------------------------------------------- conversions
__global__ void cvt_f32_bf16(const float* __restrict__ src, u16* __restrict__ dst, long n) {
  long i = (long)blockIdx.x * blockDim.x + threadIdx.x;
  const long stride = (long)gridDim.x * blockDim.x;
  for (; i < n; i += stride) dst[i] = f2bf(src[i]);
}

__global__ void cvt_strided_bf16(const float* __restrict__ src, int ldsrc, int width,
                                 u16* __restrict__ dst) {
  const long idx = (long)blockIdx.x * 256 + threadIdx.x;
  const long total = (long)TOKENS * width;
  if (idx >= total) return;
  const long row = idx / width;
  const int  col = (int)(idx - row * width);
  dst[idx] = f2bf(src[row * (long)ldsrc + col]);
}

// ---------------------------------------------------------------- RMSNorm
__global__ void rmsnorm_bf16(const float* __restrict__ x, const float* __restrict__ gamma,
                             u16* __restrict__ out) {
  const int row = blockIdx.x;
  const float* xr = x + (long)row * DIMS;
  __shared__ float red[256];
  float ss = 0.f;
  for (int i = threadIdx.x; i < DIMS; i += 256) { const float v = xr[i]; ss += v * v; }
  red[threadIdx.x] = ss;
  __syncthreads();
  for (int s = 128; s > 0; s >>= 1) {
    if ((int)threadIdx.x < s) red[threadIdx.x] += red[threadIdx.x + s];
    __syncthreads();
  }
  const float inv = rsqrtf(red[0] / (float)DIMS + 1e-6f);
  u16* orow = out + (long)row * DIMS;
  for (int i = threadIdx.x; i < DIMS; i += 256) orow[i] = f2bf(xr[i] * inv * gamma[i]);
}

// ---------------------------------------------------------------- RoPE (fp32 in, bf16 out)
__global__ void rope_bf16(const float* __restrict__ src, int ldsrc, u16* __restrict__ dst,
                          int nheads, const float* __restrict__ freqs) {
  const long idx = (long)blockIdx.x * 256 + threadIdx.x;
  const long total = (long)TOKENS * nheads * (HD / 2);
  if (idx >= total) return;
  const int  pair = (int)(idx & 63);
  const long t    = idx >> 6;
  const int  hd   = (int)(t % nheads);
  const long row  = t / nheads;
  const int  s    = (int)(row & (SEQ - 1));
  const float f = freqs[s * (HD / 2) + pair];
  float sn, c;
  __sincosf(f, &sn, &c);
  const float* p = src + row * (long)ldsrc + hd * HD + pair * 2;
  const float x0 = p[0], x1 = p[1];
  u16* q = dst + row * (long)(nheads * HD) + hd * HD + pair * 2;
  q[0] = f2bf(x0 * c - x1 * sn);
  q[1] = f2bf(x0 * sn + x1 * c);
}

// ---------------------------------------------------------------- WMMA GEMM with async-LDS weight staging
// C = A(bf16) @ W(bf16)^T (+bias)(+residual); A:[M,K], W:[N,K] row-major; block tile 32x64.
__global__ void gemm_bf16_wmma(const u16* __restrict__ A, const u16* __restrict__ W,
                               const float* __restrict__ bias, const float* __restrict__ residual,
                               float* __restrict__ C, int K, int ldC, int colOff) {
  __shared__ u16 wtile[2][64 * TWS];
  const int tid  = threadIdx.x;
  const int lane = tid & 31, wave = tid >> 5, half = lane >> 4, l16 = lane & 15;
  const int n_base = blockIdx.x * 64 + (wave & 3) * 16;
  const int m_base = blockIdx.y * 32 + (wave >> 2) * 16;
  const u16* Arow = A + (long)(m_base + l16) * K;

  // async staging: thread t copies 16 B of W: row = t/4 (0..63), quarter = t%4
  const int crow = tid >> 2, cq = tid & 3;
  const unsigned g_row_off = (unsigned)((long)(blockIdx.x * 64 + crow) * K * 2) + (unsigned)cq * 16u;
  const unsigned lds_w0 = lds_off_of(&wtile[0][crow * TWS + cq * 8]);
  const unsigned lds_w1 = lds_off_of(&wtile[1][crow * TWS + cq * 8]);
  const unsigned long long wbase = (unsigned long long)(size_t)W;

  const u32* bp0 = (const u32*)&wtile[0][((wave & 3) * 16 + l16) * TWS + half * 16];
  const u32* bp1 = (const u32*)&wtile[1][((wave & 3) * 16 + l16) * TWS + half * 16];

  f32x8 acc = {};
  const int nT = K / 32;
  async_ld_b128(lds_w0, wbase, g_row_off);  // prologue: tile 0
  for (int i = 0; i < nT; ++i) {
    const int cur = i & 1;
    if (i + 1 < nT) {
      async_ld_b128(cur ? lds_w0 : lds_w1, wbase, g_row_off + (unsigned)(i + 1) * 64u);
      asm volatile("s_wait_asynccnt 1" ::: "memory");   // tile i copy done (this wave)
    } else {
      asm volatile("s_wait_asynccnt 0" ::: "memory");
    }
    __syncthreads();                                    // all waves' copies of tile i done
    const u32* bp = cur ? bp1 : bp0;
    const int k0 = i * 32;
    FragU a, b;
#pragma unroll
    for (int v = 0; v < 8; ++v) {
      const int ka = k0 + (half << 4) + (v << 1);
      a.u[v] = *(const u32*)(Arow + ka);
      b.u[v] = bp[v];
    }
    __builtin_prefetch(Arow + k0 + 256, 0, 0);          // global_prefetch_b8 (speculative)
    acc = __builtin_amdgcn_wmma_f32_16x16x32_bf16(false, a.v, false, b.v,
                                                  (short)0, acc, false, false);
    __syncthreads();                                    // reads done before buffer reuse
  }

  const int col = n_base + l16;
  const float bv = bias ? bias[col] : 0.f;
#pragma unroll
  for (int v = 0; v < 8; ++v) {
    const int row = m_base + v + (half << 3);
    float r = acc[v] + bv;
    if (residual) r += residual[(long)row * DIMS + col];
    C[(long)row * ldC + colOff + col] = r;
  }
}

// ---------------------------------------------------------------- fused FFN W1/W3 GEMM + SiLU*mul -> bf16
__global__ void gemm_ffn13_wmma(const u16* __restrict__ A, const u16* __restrict__ W1,
                                const u16* __restrict__ W3, u16* __restrict__ G) {
  __shared__ u16 t1[2][64 * TWS];
  __shared__ u16 t3[2][64 * TWS];
  const int tid  = threadIdx.x;
  const int lane = tid & 31, wave = tid >> 5, half = lane >> 4, l16 = lane & 15;
  const int n_base = blockIdx.x * 64 + (wave & 3) * 16;
  const int m_base = blockIdx.y * 32 + (wave >> 2) * 16;
  const u16* Arow = A + (long)(m_base + l16) * DIMS;

  const int crow = tid >> 2, cq = tid & 3;
  const unsigned g_row_off = (unsigned)((long)(blockIdx.x * 64 + crow) * DIMS * 2) + (unsigned)cq * 16u;
  const unsigned lds1_0 = lds_off_of(&t1[0][crow * TWS + cq * 8]);
  const unsigned lds1_1 = lds_off_of(&t1[1][crow * TWS + cq * 8]);
  const unsigned lds3_0 = lds_off_of(&t3[0][crow * TWS + cq * 8]);
  const unsigned lds3_1 = lds_off_of(&t3[1][crow * TWS + cq * 8]);
  const unsigned long long w1b = (unsigned long long)(size_t)W1;
  const unsigned long long w3b = (unsigned long long)(size_t)W3;

  const int boff = ((wave & 3) * 16 + l16) * TWS + half * 16;
  const u32* b1p0 = (const u32*)&t1[0][boff];
  const u32* b1p1 = (const u32*)&t1[1][boff];
  const u32* b3p0 = (const u32*)&t3[0][boff];
  const u32* b3p1 = (const u32*)&t3[1][boff];

  f32x8 acc1 = {}, acc3 = {};
  const int nT = DIMS / 32;
  async_ld_b128(lds1_0, w1b, g_row_off);
  async_ld_b128(lds3_0, w3b, g_row_off);
  for (int i = 0; i < nT; ++i) {
    const int cur = i & 1;
    if (i + 1 < nT) {
      const unsigned go = g_row_off + (unsigned)(i + 1) * 64u;
      async_ld_b128(cur ? lds1_0 : lds1_1, w1b, go);
      async_ld_b128(cur ? lds3_0 : lds3_1, w3b, go);
      asm volatile("s_wait_asynccnt 2" ::: "memory");
    } else {
      asm volatile("s_wait_asynccnt 0" ::: "memory");
    }
    __syncthreads();
    const u32* b1p = cur ? b1p1 : b1p0;
    const u32* b3p = cur ? b3p1 : b3p0;
    const int k0 = i * 32;
    FragU a, b1, b3;
#pragma unroll
    for (int v = 0; v < 8; ++v) {
      const int ka = k0 + (half << 4) + (v << 1);
      a.u[v]  = *(const u32*)(Arow + ka);
      b1.u[v] = b1p[v];
      b3.u[v] = b3p[v];
    }
    __builtin_prefetch(Arow + k0 + 256, 0, 0);
    acc1 = __builtin_amdgcn_wmma_f32_16x16x32_bf16(false, a.v, false, b1.v,
                                                   (short)0, acc1, false, false);
    acc3 = __builtin_amdgcn_wmma_f32_16x16x32_bf16(false, a.v, false, b3.v,
                                                   (short)0, acc3, false, false);
    __syncthreads();
  }
  const int col = n_base + l16;
#pragma unroll
  for (int v = 0; v < 8; ++v) {
    const int row = m_base + v + (half << 3);
    const float x1 = acc1[v];
    const float g  = (x1 / (1.f + expf(-x1))) * acc3[v];
    G[(long)row * FFN + col] = f2bf(g);
  }
}

// ---------------------------------------------------------------- flash attention (WMMA, online softmax)
// one wave per (batch, head, 16-query tile); 8 waves per block.
__global__ void flash_attn_wmma(const u16* __restrict__ Q, const u16* __restrict__ Kc,
                                const u16* __restrict__ Vc, u16* __restrict__ O) {
  __shared__ u16 ldsP[8][16 * 32];  // per-wave P transpose tile (C-layout -> A-layout)
  const int lane = threadIdx.x & 31;
  const int wave = threadIdx.x >> 5;
  const int half = lane >> 4;
  const int l16  = lane & 15;
  const long gw = (long)blockIdx.x * 8 + wave;
  const int qt  = (int)(gw % (SEQ / 16));
  const int hh  = (int)((gw / (SEQ / 16)) % NHEADS);
  const int bb  = (int)(gw / ((long)(SEQ / 16) * NHEADS));
  const int kvh = hh >> 2;  // N_REP = 4

  FragU qf[4];
  {
    const u16* Qr = Q + ((long)(bb * SEQ + qt * 16 + l16)) * (NHEADS * HD) + hh * HD;
#pragma unroll
    for (int c = 0; c < 4; ++c)
#pragma unroll
      for (int v = 0; v < 8; ++v) {
        const int k = c * 32 + (half << 4) + (v << 1);
        qf[c].u[v] = *(const u32*)(Qr + k);
      }
  }

  f32x8 Oa[8];
#pragma unroll
  for (int n = 0; n < 8; ++n) Oa[n] = (f32x8){};
  float m[8], l[8];
#pragma unroll
  for (int v = 0; v < 8; ++v) { m[v] = -3.0e38f; l[v] = 0.f; }

  const float scale = 0.08838834764831845f;  // 1/sqrt(128)
  const long kvStride = NKV * HD;
  const u16* Kbase = Kc + (long)bb * SEQ * kvStride + kvh * HD;
  const u16* Vbase = Vc + (long)bb * SEQ * kvStride + kvh * HD;

  for (int kb = 0; kb < SEQ / 32; ++kb) {
    f32x8 s0 = {}, s1 = {};
    {
      const u16* Kr = Kbase + (long)(kb * 32 + l16) * kvStride;
#pragma unroll
      for (int c = 0; c < 4; ++c) {
        FragU kf;
#pragma unroll
        for (int v = 0; v < 8; ++v)
          kf.u[v] = *(const u32*)(Kr + c * 32 + half * 16 + v * 2);
        s0 = __builtin_amdgcn_wmma_f32_16x16x32_bf16(false, qf[c].v, false, kf.v,
                                                     (short)0, s0, false, false);
      }
      const u16* Kr2 = Kr + 16 * kvStride;
#pragma unroll
      for (int c = 0; c < 4; ++c) {
        FragU kf;
#pragma unroll
        for (int v = 0; v < 8; ++v)
          kf.u[v] = *(const u32*)(Kr2 + c * 32 + half * 16 + v * 2);
        s1 = __builtin_amdgcn_wmma_f32_16x16x32_bf16(false, qf[c].v, false, kf.v,
                                                     (short)0, s1, false, false);
      }
    }
    float a0[8], a1[8], bm[8];
#pragma unroll
    for (int v = 0; v < 8; ++v) {
      a0[v] = s0[v] * scale;
      a1[v] = s1[v] * scale;
      bm[v] = fmaxf(a0[v], a1[v]);
    }
#pragma unroll
    for (int v = 0; v < 8; ++v) {
      bm[v] = fmaxf(bm[v], __shfl_xor(bm[v], 1, 32));
      bm[v] = fmaxf(bm[v], __shfl_xor(bm[v], 2, 32));
      bm[v] = fmaxf(bm[v], __shfl_xor(bm[v], 4, 32));
      bm[v] = fmaxf(bm[v], __shfl_xor(bm[v], 8, 32));
    }
    float alpha[8], rs[8];
#pragma unroll
    for (int v = 0; v < 8; ++v) {
      const float mn = fmaxf(m[v], bm[v]);
      alpha[v] = expf(m[v] - mn);
      m[v] = mn;
      a0[v] = expf(a0[v] - mn);
      a1[v] = expf(a1[v] - mn);
      rs[v] = a0[v] + a1[v];
    }
#pragma unroll
    for (int v = 0; v < 8; ++v) {
      rs[v] += __shfl_xor(rs[v], 1, 32);
      rs[v] += __shfl_xor(rs[v], 2, 32);
      rs[v] += __shfl_xor(rs[v], 4, 32);
      rs[v] += __shfl_xor(rs[v], 8, 32);
      l[v] = l[v] * alpha[v] + rs[v];
    }
#pragma unroll
    for (int n = 0; n < 8; ++n)
#pragma unroll
      for (int v = 0; v < 8; ++v) Oa[n][v] *= alpha[v];

    u16* Pw = &ldsP[wave][0];
#pragma unroll
    for (int v = 0; v < 8; ++v) {
      const int prow = v + 8 * half;
      Pw[prow * 32 + l16]      = f2bf(a0[v]);
      Pw[prow * 32 + 16 + l16] = f2bf(a1[v]);
    }
    asm volatile("s_wait_dscnt 0x0" ::: "memory");
    FragU pf;
#pragma unroll
    for (int v = 0; v < 8; ++v) {
      const int pk = (half << 4) + (v << 1);
      pf.u[v] = *(const u32*)(Pw + l16 * 32 + pk);
    }

#pragma unroll
    for (int n = 0; n < 8; ++n) {
      FragU vf;
      const u16* Vd = Vbase + n * 16 + l16;
#pragma unroll
      for (int v = 0; v < 8; ++v) {
        const long key = (long)kb * 32 + half * 16 + v * 2;
        const u32 e0 = Vd[key * kvStride];
        const u32 e1 = Vd[(key + 1) * kvStride];
        vf.u[v] = e0 | (e1 << 16);
      }
      Oa[n] = __builtin_amdgcn_wmma_f32_16x16x32_bf16(false, pf.v, false, vf.v,
                                                      (short)0, Oa[n], false, false);
    }
  }

  u16* Orow0 = O + ((long)(bb * SEQ + qt * 16)) * DIMS + hh * HD;
#pragma unroll
  for (int v = 0; v < 8; ++v) {
    const float invl = 1.f / l[v];
    const int row = v + 8 * half;
#pragma unroll
    for (int n = 0; n < 8; ++n)
      Orow0[(long)row * DIMS + n * 16 + l16] = f2bf(Oa[n][v] * invl);
  }
}

// ---------------------------------------------------------------- host orchestration
extern "C" void kernel_launch(void* const* d_in, const int* in_sizes, int n_in,
                              void* d_out, int out_size, void* d_ws, size_t ws_size,
                              hipStream_t stream) {
  (void)in_sizes; (void)n_in; (void)out_size; (void)ws_size;
  const float* x      = (const float*)d_in[0];
  const float* freqs  = (const float*)d_in[1];
  const float* g_attn = (const float*)d_in[3];
  const float* g_ffn  = (const float*)d_in[4];
  const float* wq_w = (const float*)d_in[5];  const float* wq_b = (const float*)d_in[6];
  const float* wk_w = (const float*)d_in[7];  const float* wk_b = (const float*)d_in[8];
  const float* wv_w = (const float*)d_in[9];  const float* wv_b = (const float*)d_in[10];
  const float* wo_w = (const float*)d_in[11]; const float* wo_b = (const float*)d_in[12];
  const float* w1_w = (const float*)d_in[13];
  const float* w3_w = (const float*)d_in[14];
  const float* w2_w = (const float*)d_in[15];
  float* out = (float*)d_out;

  char* ws = (char*)d_ws;
  size_t off = 0;
  auto alloc = [&](size_t bytes) {
    char* p = ws + off;
    off += (bytes + 255) & ~(size_t)255;
    return p;
  };
  u16* wq_bf = (u16*)alloc((size_t)DIMS * DIMS * 2);
  u16* wk_bf = (u16*)alloc((size_t)(NKV * HD) * DIMS * 2);
  u16* wv_bf = (u16*)alloc((size_t)(NKV * HD) * DIMS * 2);
  u16* wo_bf = (u16*)alloc((size_t)DIMS * DIMS * 2);
  u16* w1_bf = (u16*)alloc((size_t)FFN * DIMS * 2);
  u16* w3_bf = (u16*)alloc((size_t)FFN * DIMS * 2);
  u16* w2_bf = (u16*)alloc((size_t)DIMS * FFN * 2);
  u16* xn_bf = (u16*)alloc((size_t)TOKENS * DIMS * 2);          // also reused as hn
  float* qkv = (float*)alloc((size_t)TOKENS * 6144 * 4);        // fp32 QKV (192 MiB)
  u16* attn_bf = (u16*)qkv;   // aliases qkv (qkv dead after RoPE/cvt)
  u16* g_bf    = (u16*)qkv;   // aliases qkv (attn dead after wo-gemm)
  u16* q_bf = (u16*)alloc((size_t)TOKENS * DIMS * 2);
  u16* k_bf = (u16*)alloc((size_t)TOKENS * NKV * HD * 2);
  u16* v_bf = (u16*)alloc((size_t)TOKENS * NKV * HD * 2);
  float* h  = (float*)alloc((size_t)TOKENS * DIMS * 4);

  // weights -> bf16 (one streaming pass over 708 MB of fp32)
  cvt_f32_bf16<<<4096, 256, 0, stream>>>(wq_w, wq_bf, (long)DIMS * DIMS);
  cvt_f32_bf16<<<4096, 256, 0, stream>>>(wk_w, wk_bf, (long)(NKV * HD) * DIMS);
  cvt_f32_bf16<<<4096, 256, 0, stream>>>(wv_w, wv_bf, (long)(NKV * HD) * DIMS);
  cvt_f32_bf16<<<4096, 256, 0, stream>>>(wo_w, wo_bf, (long)DIMS * DIMS);
  cvt_f32_bf16<<<8192, 256, 0, stream>>>(w1_w, w1_bf, (long)FFN * DIMS);
  cvt_f32_bf16<<<8192, 256, 0, stream>>>(w3_w, w3_bf, (long)FFN * DIMS);
  cvt_f32_bf16<<<8192, 256, 0, stream>>>(w2_w, w2_bf, (long)DIMS * FFN);

  const dim3 blk(256);
  rmsnorm_bf16<<<TOKENS, 256, 0, stream>>>(x, g_attn, xn_bf);
  gemm_bf16_wmma<<<dim3(DIMS / 64, TOKENS / 32), blk, 0, stream>>>(
      xn_bf, wq_bf, wq_b, nullptr, qkv, DIMS, 6144, 0);
  gemm_bf16_wmma<<<dim3((NKV * HD) / 64, TOKENS / 32), blk, 0, stream>>>(
      xn_bf, wk_bf, wk_b, nullptr, qkv, DIMS, 6144, 4096);
  gemm_bf16_wmma<<<dim3((NKV * HD) / 64, TOKENS / 32), blk, 0, stream>>>(
      xn_bf, wv_bf, wv_b, nullptr, qkv, DIMS, 6144, 5120);
  rope_bf16<<<(TOKENS * NHEADS * (HD / 2)) / 256, 256, 0, stream>>>(qkv, 6144, q_bf, NHEADS, freqs);
  rope_bf16<<<(TOKENS * NKV * (HD / 2)) / 256, 256, 0, stream>>>(qkv + 4096, 6144, k_bf, NKV, freqs);
  cvt_strided_bf16<<<(TOKENS * NKV * HD) / 256, 256, 0, stream>>>(qkv + 5120, 6144, NKV * HD, v_bf);
  flash_attn_wmma<<<(BATCH * NHEADS * (SEQ / 16)) / 8, 256, 0, stream>>>(q_bf, k_bf, v_bf, attn_bf);
  gemm_bf16_wmma<<<dim3(DIMS / 64, TOKENS / 32), blk, 0, stream>>>(
      attn_bf, wo_bf, wo_b, x, h, DIMS, DIMS, 0);
  rmsnorm_bf16<<<TOKENS, 256, 0, stream>>>(h, g_ffn, xn_bf);
  gemm_ffn13_wmma<<<dim3(FFN / 64, TOKENS / 32), blk, 0, stream>>>(xn_bf, w1_bf, w3_bf, g_bf);
  gemm_bf16_wmma<<<dim3(DIMS / 64, TOKENS / 32), blk, 0, stream>>>(
      g_bf, w2_bf, nullptr, h, out, FFN, DIMS, 0);
}
